// CrossTransformerBlock_18399639896318
// MI455X (gfx1250) — matrix operand order repaired and checked by reading, not compiled, not run on hardware
//
#include <hip/hip_runtime.h>
#include <math.h>

typedef __attribute__((ext_vector_type(16))) _Float16 v16h;
typedef __attribute__((ext_vector_type(8)))  _Float16 v8h;
typedef __attribute__((ext_vector_type(8)))  float    v8f;
typedef __attribute__((ext_vector_type(4)))  unsigned int u32x4;
typedef __attribute__((ext_vector_type(8)))  int      i32x8;
typedef __attribute__((ext_vector_type(4)))  int      i32x4;

#if defined(__gfx1250__) && __has_builtin(__builtin_amdgcn_tensor_load_to_lds) && \
    __has_builtin(__builtin_amdgcn_s_wait_tensorcnt)
#define USE_TDM 1
#endif

union V16 { v16h v; _Float16 e[16]; };

__device__ __forceinline__ v16h cat16(v8h a, v8h b) {
    return __builtin_shufflevector(a, b, 0,1,2,3,4,5,6,7,8,9,10,11,12,13,14,15);
}
__device__ __forceinline__ v8h ld8(const _Float16* p) { return *(const v8h*)p; }

__device__ __forceinline__ v8f wmma_f16(v16h a, v16h b, v8f c) {
    // D(f32,16x16) = A(f16,16x32) x B(f16,32x16) + C
    return __builtin_amdgcn_wmma_f32_16x16x32_f16(false, a, false, b, (short)0, c, false, false);
}

#ifdef USE_TDM
// TDM: DMA a [rows x 32-half] f16 tile (row stride K halves in memory) into LDS,
// padding +4 DWORDs per 16-DWORD row -> LDS row stride = 40 halves.
__device__ __forceinline__ void tdm_load_tile(unsigned lds_off, const _Float16* gptr,
                                              int K, int rows) {
    unsigned long long ga = (unsigned long long)(uintptr_t)gptr;
    u32x4 g0;
    g0[0] = 1u;                                               // count=1, user mode
    g0[1] = lds_off;                                          // lds_addr (bytes)
    g0[2] = (unsigned)ga;                                     // global_addr[31:0]
    g0[3] = (unsigned)((ga >> 32) & 0x01FFFFFFu) | (2u << 30);// addr[56:32] | type=2
    i32x8 g1;
    g1[0] = (1 << 16)   // data_size = 2 bytes
          | (1 << 20)   // pad_enable
          | (3 << 22)   // pad_interval: 16 DWORDs (one 32-half row)
          | (3 << 25);  // pad_amount: 4 DWORDs (16B) -> stride 40 halves
    g1[1] = (K & 0xFFFF) << 16;                               // tensor_dim0[15:0]
    g1[2] = ((K >> 16) & 0xFFFF) | ((rows & 0xFFFF) << 16);   // dim0[31:16] | tensor_dim1[15:0]
    g1[3] = (32 & 0xFFFF) << 16;                              // tensor_dim1[31:16]=0 | tile_dim0=32
    g1[4] = rows & 0xFFFF;                                    // tile_dim1=rows | tile_dim2=0
    g1[5] = K;                                                // tensor_dim0_stride[31:0]
    g1[6] = 0;
    g1[7] = 0;
    i32x4 z4 = {0, 0, 0, 0};
    i32x8 z8 = {0, 0, 0, 0, 0, 0, 0, 0};
    // clang-23 / therock form: (g0, g1, g2, g3, extra, cpol)
    __builtin_amdgcn_tensor_load_to_lds(g0, g1, z4, z4, z8, 0);
}
#endif

// ---------------------------------------------------------------------------
// f32 -> f16 conversion (grid-stride)
// ---------------------------------------------------------------------------
__global__ void cvt_kernel(const float* __restrict__ src, _Float16* __restrict__ dst, size_t n) {
    size_t i = (size_t)blockIdx.x * blockDim.x + threadIdx.x;
    size_t stride = (size_t)gridDim.x * blockDim.x;
    for (; i < n; i += stride) dst[i] = (_Float16)src[i];
}

// qkv f32 [M, d3] -> f16, scaling the Q block (cols < D) by 1/sqrt(dh)
__global__ void cvt_qkv_kernel(const float* __restrict__ src, _Float16* __restrict__ dst,
                               size_t n, int d3, int D) {
    size_t i = (size_t)blockIdx.x * blockDim.x + threadIdx.x;
    size_t stride = (size_t)gridDim.x * blockDim.x;
    for (; i < n; i += stride) {
        int col = (int)(i % (size_t)d3);
        float s = (col < D) ? 0.125f : 1.0f;
        dst[i] = (_Float16)(src[i] * s);
    }
}

// V transpose: vT[((b*H+h)*64+c)*S + t] = qkvh[(b*S+t)*d3 + voff + h*64 + c]
__global__ void vtrans_kernel(const _Float16* __restrict__ qkvh, _Float16* __restrict__ vT,
                              size_t n, int S, int d3, int voff, int H) {
    size_t i = (size_t)blockIdx.x * blockDim.x + threadIdx.x;
    size_t stride = (size_t)gridDim.x * blockDim.x;
    for (; i < n; i += stride) {
        int t = (int)(i % (size_t)S);
        size_t rest = i / (size_t)S;
        int c = (int)(rest % 64);
        size_t bh = rest / 64;
        int h = (int)(bh % (size_t)H);
        size_t b = bh / (size_t)H;
        vT[i] = qkvh[(b * S + t) * (size_t)d3 + voff + h * 64 + c];
    }
}

// ---------------------------------------------------------------------------
// WMMA GEMM: C[M,N] = A[M,K](f16) * W[N,K]^T(f16) + bias[N]
// Block tile 128(M) x 64(N), 8 waves, wave tile 32x32 (2x2 WMMA), K-step 32.
// Tiles staged by the Tensor Data Mover (double-buffered) when available.
// MODE 0: write f32 C.   MODE 1: exact-erf GELU, write f16 C.
// ---------------------------------------------------------------------------
template <int MODE>
__global__ __launch_bounds__(256)
void gemm_kernel(const _Float16* __restrict__ A, const _Float16* __restrict__ W,
                 const float* __restrict__ bias, float* __restrict__ Cf,
                 _Float16* __restrict__ Ch, int K, int ldc) {
    __shared__ __align__(16) _Float16 As[2 * 128 * 40];  // stride 40 halves/row
    __shared__ __align__(16) _Float16 Ws[2 * 64 * 40];

    const int tid  = threadIdx.x;
    const int lane = tid & 31;
    const int w    = tid >> 5;
    const int wm   = w & 3;        // 4 waves along M
    const int wn   = w >> 2;       // 2 waves along N
    const int r    = lane & 15;
    const int hi   = lane >> 4;
    const int m0   = blockIdx.y * 128;
    const int n0   = blockIdx.x * 64;

    v8f c00 = {}, c01 = {}, c10 = {}, c11 = {};
    int buf = 0;

#ifdef USE_TDM
    const unsigned asBase = (unsigned)(uintptr_t)(void*)&As[0];
    const unsigned wsBase = (unsigned)(uintptr_t)(void*)&Ws[0];
    const unsigned ASZ = 128 * 40 * 2, WSZ = 64 * 40 * 2;
    if (w == 0) {
        tdm_load_tile(asBase, &A[(size_t)m0 * K], K, 128);
        tdm_load_tile(wsBase, &W[(size_t)n0 * K], K, 64);
    }
#endif

    for (int k0 = 0; k0 < K; k0 += 32) {
#ifdef USE_TDM
        if (w == 0) {
            if (k0 + 32 < K) {  // prefetch next tile pair into the other buffer
                tdm_load_tile(asBase + (buf ^ 1) * ASZ, &A[(size_t)m0 * K + k0 + 32], K, 128);
                tdm_load_tile(wsBase + (buf ^ 1) * WSZ, &W[(size_t)n0 * K + k0 + 32], K, 64);
                __builtin_amdgcn_s_wait_tensorcnt(2);  // current pair complete
            } else {
                __builtin_amdgcn_s_wait_tensorcnt(0);
            }
        }
#else
        {   // manual staging into buffer 0
#pragma unroll
            for (int i = 0; i < 2; ++i) {
                int c = tid + i * 256;
                int arow = c >> 2, kc = (c & 3) * 8;
                *(float4*)&As[arow * 40 + kc] =
                    *(const float4*)&A[(size_t)(m0 + arow) * K + k0 + kc];
            }
            int wrow = tid >> 2, kc = (tid & 3) * 8;
            *(float4*)&Ws[wrow * 40 + kc] =
                *(const float4*)&W[(size_t)(n0 + wrow) * K + k0 + kc];
        }
#endif
        __syncthreads();

        const _Float16* Ab = &As[buf * 128 * 40];
        const _Float16* Wb = &Ws[buf * 64 * 40];
        // A frag: lane r = row, elem e -> K = hi*16 + e
        v16h a0 = cat16(ld8(&Ab[(wm * 32 +      r) * 40 + hi * 16]),
                        ld8(&Ab[(wm * 32 +      r) * 40 + hi * 16 + 8]));
        v16h a1 = cat16(ld8(&Ab[(wm * 32 + 16 + r) * 40 + hi * 16]),
                        ld8(&Ab[(wm * 32 + 16 + r) * 40 + hi * 16 + 8]));
        // B frag: lane r = col N, elem e -> K = hi*16 + e
        const int kb = hi * 16;
        v16h b0 = cat16(ld8(&Wb[(wn * 32 +      r) * 40 + kb]),
                        ld8(&Wb[(wn * 32 +      r) * 40 + kb + 8]));
        v16h b1 = cat16(ld8(&Wb[(wn * 32 + 16 + r) * 40 + kb]),
                        ld8(&Wb[(wn * 32 + 16 + r) * 40 + kb + 8]));

        c00 = wmma_f16(a0, b0, c00);
        c01 = wmma_f16(a0, b1, c01);
        c10 = wmma_f16(a1, b0, c10);
        c11 = wmma_f16(a1, b1, c11);
        __syncthreads();
#ifdef USE_TDM
        buf ^= 1;
#endif
    }

    // Epilogue: C element v -> row = v + hi*8, col = lane&15
    const int nc = lane & 15;
    v8f* accs[4] = { &c00, &c01, &c10, &c11 };
#pragma unroll
    for (int q = 0; q < 4; ++q) {
        const int ms = (q >> 1) * 16, ns = (q & 1) * 16;
#pragma unroll
        for (int v = 0; v < 8; ++v) {
            int gm = m0 + wm * 32 + ms + hi * 8 + v;
            int gn = n0 + wn * 32 + ns + nc;
            float t = (*accs[q])[v] + bias[gn];
            if (MODE == 1) {
                t = 0.5f * t * (1.0f + erff(t * 0.70710678118654752f));  // exact GELU
                Ch[(size_t)gm * ldc + gn] = (_Float16)t;
            } else {
                Cf[(size_t)gm * ldc + gn] = t;
            }
        }
    }
}

// ---------------------------------------------------------------------------
// Flash-style attention, all-f16 operands. One wave per (b, h, 16-row S tile).
// qkvh: f16 [B,S,d3] (Q pre-scaled by 1/sqrt(dh)); vT: f16 [B,H,64,S].
// Pass 1: lane-local running row max (no shuffles); one cross-lane reduce.
// Pass 2: recompute scores, p=exp(sc-max) -> LDS f16, lane-local row-sum,
//         unnormalized O += P*V (WMMA); one cross-lane sum reduce at the end,
//         then scale O by 1/rowsum before store. Exact softmax.
// ---------------------------------------------------------------------------
__global__ __launch_bounds__(32)
void attn_kernel(const _Float16* __restrict__ qkvh, const _Float16* __restrict__ vT,
                 _Float16* __restrict__ outh, int S, int d3, int koff, int H, int dmodel) {
    __shared__ __align__(16) _Float16 pt[16 * 40];

    const int lane = threadIdx.x & 31;
    const int nl   = lane & 15;
    const int hi   = lane >> 4;
    const int st   = blockIdx.x;
    const int h    = blockIdx.y;
    const int b    = blockIdx.z;
    const int hb   = h * 64;

    // Q fragments: two 16x32 K-chunks, each two contiguous 8-half vectors
    const size_t qrow = ((size_t)b * S + st * 16 + nl) * d3 + hb;
    v16h aq0 = cat16(ld8(&qkvh[qrow + hi * 16]),      ld8(&qkvh[qrow + hi * 16 + 8]));
    v16h aq1 = cat16(ld8(&qkvh[qrow + 32 + hi * 16]), ld8(&qkvh[qrow + 32 + hi * 16 + 8]));

    // ---- Pass 1: lane-local row max over all score tiles (no cross-lane ops) ----
    float rmax[8];
#pragma unroll
    for (int v = 0; v < 8; ++v) rmax[v] = -1e30f;

    for (int tt = 0; tt < S / 16; ++tt) {
        const size_t trow = ((size_t)b * S + tt * 16 + nl) * d3 + koff + hb;
        v16h bk0 = cat16(ld8(&qkvh[trow + hi * 16]),      ld8(&qkvh[trow + hi * 16 + 8]));
        v16h bk1 = cat16(ld8(&qkvh[trow + 32 + hi * 16]), ld8(&qkvh[trow + 32 + hi * 16 + 8]));
        v8f sc = {};
        sc = wmma_f16(aq0, bk0, sc);
        sc = wmma_f16(aq1, bk1, sc);
#pragma unroll
        for (int v = 0; v < 8; ++v) rmax[v] = fmaxf(rmax[v], sc[v]);
    }
    // One cross-lane max reduction per row (within each 16-lane group)
#pragma unroll
    for (int v = 0; v < 8; ++v) {
        float tv = rmax[v];
        tv = fmaxf(tv, __shfl_xor(tv, 1, 32));
        tv = fmaxf(tv, __shfl_xor(tv, 2, 32));
        tv = fmaxf(tv, __shfl_xor(tv, 4, 32));
        tv = fmaxf(tv, __shfl_xor(tv, 8, 32));
        rmax[v] = tv;
    }

    // ---- Pass 2: unnormalized P and O, lane-local row sums ----
    float rsum[8];
#pragma unroll
    for (int v = 0; v < 8; ++v) rsum[v] = 0.0f;

    v8f o0 = {}, o1 = {}, o2 = {}, o3 = {};
    for (int kk = 0; kk < S / 32; ++kk) {
#pragma unroll
        for (int sub = 0; sub < 2; ++sub) {
            int tt = kk * 2 + sub;
            const size_t trow = ((size_t)b * S + tt * 16 + nl) * d3 + koff + hb;
            v16h bk0 = cat16(ld8(&qkvh[trow + hi * 16]),      ld8(&qkvh[trow + hi * 16 + 8]));
            v16h bk1 = cat16(ld8(&qkvh[trow + 32 + hi * 16]), ld8(&qkvh[trow + 32 + hi * 16 + 8]));
            v8f sc = {};
            sc = wmma_f16(aq0, bk0, sc);
            sc = wmma_f16(aq1, bk1, sc);
#pragma unroll
            for (int v = 0; v < 8; ++v) {
                float p = __expf(sc[v] - rmax[v]);   // <= 1, safe in f16
                rsum[v] += p;
                pt[(hi * 8 + v) * 40 + sub * 16 + nl] = (_Float16)p;  // P tile 16x32
            }
        }
        // A fragment of P from LDS (same-wave LDS ops are in program order)
        v16h ap = cat16(ld8(&pt[nl * 40 + hi * 16]), ld8(&pt[nl * 40 + hi * 16 + 8]));
        v8f* oa[4] = { &o0, &o1, &o2, &o3 };
#pragma unroll
        for (int n4 = 0; n4 < 4; ++n4) {
            // V B-fragment: lane col = n4*16+nl (dh row of vT), elem e -> t contiguous
            const size_t vrow = (((size_t)b * H + h) * 64 + n4 * 16 + nl) * S + kk * 32 + hi * 16;
            v16h bv = cat16(ld8(&vT[vrow]), ld8(&vT[vrow + 8]));
            *oa[n4] = wmma_f16(ap, bv, *oa[n4]);
        }
    }

    // One cross-lane sum reduction per row, then invert
    float rinv[8];
#pragma unroll
    for (int v = 0; v < 8; ++v) {
        float sv = rsum[v];
        sv += __shfl_xor(sv, 1, 32);
        sv += __shfl_xor(sv, 2, 32);
        sv += __shfl_xor(sv, 4, 32);
        sv += __shfl_xor(sv, 8, 32);
        rinv[v] = 1.0f / sv;
    }

    // Store O (f16), normalized per row
    v8f* oa[4] = { &o0, &o1, &o2, &o3 };
#pragma unroll
    for (int n4 = 0; n4 < 4; ++n4) {
#pragma unroll
        for (int v = 0; v < 8; ++v) {
            int s = st * 16 + hi * 8 + v;
            outh[((size_t)b * S + s) * dmodel + hb + n4 * 16 + nl] =
                (_Float16)((*oa[n4])[v] * rinv[v]);
        }
    }
}

// ---------------------------------------------------------------------------
// Fused residual + LayerNorm. y = LN(a + res)*g + b.  Writes f32 y, optional f16.
// ---------------------------------------------------------------------------
__global__ __launch_bounds__(256)
void ln_kernel(const float* __restrict__ a, const float* __restrict__ res,
               const float* __restrict__ g, const float* __restrict__ bt,
               float* __restrict__ y, _Float16* __restrict__ yh, int d) {
    const int row = blockIdx.x;
    const size_t base = (size_t)row * d;
    __shared__ float s1[256], s2[256];
    float ls = 0.0f, lq = 0.0f;
    for (int i = threadIdx.x; i < d; i += 256) {
        float v = a[base + i] + res[base + i];
        ls += v; lq += v * v;
    }
    s1[threadIdx.x] = ls; s2[threadIdx.x] = lq;
    __syncthreads();
    for (int stp = 128; stp > 0; stp >>= 1) {
        if (threadIdx.x < stp) {
            s1[threadIdx.x] += s1[threadIdx.x + stp];
            s2[threadIdx.x] += s2[threadIdx.x + stp];
        }
        __syncthreads();
    }
    float mu   = s1[0] / d;
    float var  = s2[0] / d - mu * mu;
    float rstd = rsqrtf(var + 1e-5f);
    for (int i = threadIdx.x; i < d; i += 256) {
        float v = a[base + i] + res[base + i];
        float o = (v - mu) * rstd * g[i] + bt[i];
        y[base + i] = o;
        if (yh) yh[base + i] = (_Float16)o;
    }
}

// ---------------------------------------------------------------------------
// Host orchestration
// ---------------------------------------------------------------------------
extern "C" void kernel_launch(void* const* d_in, const int* in_sizes, int n_in,
                              void* d_out, int out_size, void* d_ws, size_t ws_size,
                              hipStream_t stream) {
    (void)in_sizes; (void)n_in; (void)out_size; (void)ws_size;
    const int B = 4, S = 1024, D = 1024, H = 16, FF = 4096;
    const int M = B * S;

    const float* x        = (const float*)d_in[0];
    const float* mem      = (const float*)d_in[1];
    const float* sa_in_w  = (const float*)d_in[2];
    const float* sa_in_b  = (const float*)d_in[3];
    const float* sa_out_w = (const float*)d_in[4];
    const float* sa_out_b = (const float*)d_in[5];
    const float* ca_in_w  = (const float*)d_in[6];
    const float* ca_in_b  = (const float*)d_in[7];
    const float* ca_out_w = (const float*)d_in[8];
    const float* ca_out_b = (const float*)d_in[9];
    const float* ff_w1    = (const float*)d_in[10];
    const float* ff_b1    = (const float*)d_in[11];
    const float* ff_w2    = (const float*)d_in[12];
    const float* ff_b2    = (const float*)d_in[13];
    const float* ln1_g    = (const float*)d_in[14];
    const float* ln1_b    = (const float*)d_in[15];
    const float* ln2_g    = (const float*)d_in[16];
    const float* ln2_b    = (const float*)d_in[17];
    const float* ln3_g    = (const float*)d_in[18];
    const float* ln3_b    = (const float*)d_in[19];
    float* out = (float*)d_out;

    // Workspace partition (256B aligned)
    char* wsp = (char*)d_ws;
    size_t off = 0;
    auto alloc = [&](size_t bytes) -> void* {
        void* p = wsp + off;
        off = (off + bytes + 255) & ~(size_t)255;
        return p;
    };
    _Float16* wsa_in_h  = (_Float16*)alloc((size_t)3 * D * D * 2);
    _Float16* wsa_out_h = (_Float16*)alloc((size_t)D * D * 2);
    _Float16* wca_in_h  = (_Float16*)alloc((size_t)3 * D * D * 2);
    _Float16* wca_out_h = (_Float16*)alloc((size_t)D * D * 2);
    _Float16* wff1_h    = (_Float16*)alloc((size_t)FF * D * 2);
    _Float16* wff2_h    = (_Float16*)alloc((size_t)FF * D * 2);
    _Float16* xh        = (_Float16*)alloc((size_t)M * D * 2);
    _Float16* memh      = (_Float16*)alloc((size_t)M * D * 2);
    float*    qkvb      = (float*)   alloc((size_t)M * 3 * D * 4);
    _Float16* qkvh      = (_Float16*)alloc((size_t)M * 3 * D * 2);
    _Float16* vTb       = (_Float16*)alloc((size_t)M * D * 2);
    _Float16* attnoh    = (_Float16*)alloc((size_t)M * D * 2);
    float*    proj      = (float*)   alloc((size_t)M * D * 4);
    float*    xa        = (float*)   alloc((size_t)M * D * 4);
    _Float16* xah       = (_Float16*)alloc((size_t)M * D * 2);
    float*    xb        = (float*)   alloc((size_t)M * D * 4);
    _Float16* xbh       = (_Float16*)alloc((size_t)M * D * 2);
    _Float16* hh        = (_Float16*)alloc((size_t)M * FF * 2);

    auto cvt = [&](const float* s, _Float16* dst, size_t n) {
        cvt_kernel<<<dim3(2048), 256, 0, stream>>>(s, dst, n);
    };
    auto prep_attn = [&]() {  // qkvb(f32) -> qkvh(f16, Q scaled) -> vT
        cvt_qkv_kernel<<<dim3(2048), 256, 0, stream>>>(qkvb, qkvh, (size_t)M * 3 * D, 3 * D, D);
        vtrans_kernel<<<dim3(2048), 256, 0, stream>>>(qkvh, vTb, (size_t)M * D, S, 3 * D, 2 * D, H);
    };

    // f32 -> f16 staging
    cvt(sa_in_w,  wsa_in_h,  (size_t)3 * D * D);
    cvt(sa_out_w, wsa_out_h, (size_t)D * D);
    cvt(ca_in_w,  wca_in_h,  (size_t)3 * D * D);
    cvt(ca_out_w, wca_out_h, (size_t)D * D);
    cvt(ff_w1,    wff1_h,    (size_t)FF * D);
    cvt(ff_w2,    wff2_h,    (size_t)FF * D);
    cvt(x,        xh,        (size_t)M * D);
    cvt(mem,      memh,      (size_t)M * D);

    const dim3 gblk(256);
    // ---- Self attention ----
    gemm_kernel<0><<<dim3(3 * D / 64, M / 128), gblk, 0, stream>>>(
        xh, wsa_in_h, sa_in_b, qkvb, nullptr, D, 3 * D);
    prep_attn();
    attn_kernel<<<dim3(S / 16, H, B), 32, 0, stream>>>(
        qkvh, vTb, attnoh, S, 3 * D, D, H, D);
    gemm_kernel<0><<<dim3(D / 64, M / 128), gblk, 0, stream>>>(
        attnoh, wsa_out_h, sa_out_b, proj, nullptr, D, D);
    ln_kernel<<<M, 256, 0, stream>>>(x, proj, ln1_g, ln1_b, xa, xah, D);

    // ---- Cross attention (Q from xa, K/V from mem) ----
    gemm_kernel<0><<<dim3(D / 64, M / 128), gblk, 0, stream>>>(
        xah, wca_in_h, ca_in_b, qkvb, nullptr, D, 3 * D);
    gemm_kernel<0><<<dim3(2 * D / 64, M / 128), gblk, 0, stream>>>(
        memh, wca_in_h + (size_t)D * D, ca_in_b + D, qkvb + D, nullptr, D, 3 * D);
    prep_attn();
    attn_kernel<<<dim3(S / 16, H, B), 32, 0, stream>>>(
        qkvh, vTb, attnoh, S, 3 * D, D, H, D);
    gemm_kernel<0><<<dim3(D / 64, M / 128), gblk, 0, stream>>>(
        attnoh, wca_out_h, ca_out_b, proj, nullptr, D, D);
    ln_kernel<<<M, 256, 0, stream>>>(xa, proj, ln2_g, ln2_b, xb, xbh, D);

    // ---- FFN (GEMM1 fuses bias + exact GELU, f16 out) ----
    gemm_kernel<1><<<dim3(FF / 64, M / 128), gblk, 0, stream>>>(
        xbh, wff1_h, ff_b1, nullptr, hh, D, FF);
    gemm_kernel<0><<<dim3(D / 64, M / 128), gblk, 0, stream>>>(
        hh, wff2_h, ff_b2, proj, nullptr, FF, D);
    ln_kernel<<<M, 256, 0, stream>>>(xb, proj, ln3_g, ln3_b, out, nullptr, D);
}
